// HCMGNN_Layer_74440373174626
// MI455X (gfx1250) — compile-verified, matrix operand order checked by reading, not run on hardware
//
#include <hip/hip_runtime.h>
#include <hip/hip_bf16.h>
#include <math.h>

typedef _Float16 half16 __attribute__((ext_vector_type(16)));
typedef float    f8     __attribute__((ext_vector_type(8)));

#define E_EDGES 250000
#define DDIM 64
#define HHEADS 8
#define NG 20000
#define NM 30000
#define ND 10000
#define NTOT 60000
#define FDIM 512            /* H*D */
#define HID 128
#define LALPHA 0.01f

/* workspace layout in floats */
#define OFF_EMB 0L                       /* E*64 = 16,000,000 */
#define OFF_A   16000000L                /* E*8  =  2,000,000 */
#define OFF_A1  18000000L                /* 60000*8 = 480,000 */
#define OFF_M   18480000L                /* 480,000 */
#define OFF_DEN 18960000L                /* 480,000 */
#define OFF_W1H 19440000L                /* 65536 halves = 32768 floats */
#define OFF_SC  19473024L                /* 6 floats */
#define OFF_Z1  19480000L                /* 60000*512 = 30,720,000 */
#define ZTOT    30720000L

__global__ void fill_kernel(float* __restrict__ p, float v, long n) {
  long i = (long)blockIdx.x * blockDim.x + threadIdx.x;
  if (i < n) p[i] = v;
}

/* a1[node,h] = dot(feats[node], attn1[h])  for all three node types concatenated */
__global__ void a1_kernel(const float* __restrict__ hg, const float* __restrict__ hm,
                          const float* __restrict__ hd,
                          const float* __restrict__ wg, const float* __restrict__ wm,
                          const float* __restrict__ wd, float* __restrict__ a1) {
  int id = blockIdx.x * blockDim.x + threadIdx.x;
  if (id >= NTOT * HHEADS) return;
  int node = id >> 3, h = id & 7;
  const float* f; const float* w; int ln;
  if (node < NG)            { f = hg; w = wg; ln = node; }
  else if (node < NG + NM)  { f = hm; w = wm; ln = node - NG; }
  else                      { f = hd; w = wd; ln = node - NG - NM; }
  float acc = 0.f;
  #pragma unroll 8
  for (int d = 0; d < DDIM; ++d) acc += f[(long)ln * DDIM + d] * w[h * DDIM + d];
  a1[id] = acc;
}

/* emb[e] = (h_g[ig] + conj(norm(rv_p)) * h_m[im] + h_d[id]) / 3 ; one thread per complex pair */
__global__ void semantic_kernel(const float* __restrict__ hg, const float* __restrict__ hm,
                                const float* __restrict__ hd,
                                const int* __restrict__ ig, const int* __restrict__ im,
                                const int* __restrict__ idd,
                                const float* __restrict__ rvec, int path,
                                float* __restrict__ emb) {
  long t = (long)blockIdx.x * blockDim.x + threadIdx.x;
  if (t >= (long)E_EDGES * 32) return;
  int e = (int)(t >> 5), k = (int)(t & 31);
  float c0 = rvec[path * 64 + 2 * k], d0 = rvec[path * 64 + 2 * k + 1];
  float inv = rsqrtf(c0 * c0 + d0 * d0);
  float c = c0 * inv, d = -d0 * inv;            /* conjugate of normalized rv */
  int giN = ig[e], miN = im[e], diN = idd[e];
  float g0 = hg[(long)giN * 64 + 2 * k], g1 = hg[(long)giN * 64 + 2 * k + 1];
  float m0 = hm[(long)miN * 64 + 2 * k], m1 = hm[(long)miN * 64 + 2 * k + 1];
  float x0 = hd[(long)diN * 64 + 2 * k], x1 = hd[(long)diN * 64 + 2 * k + 1];
  float re = m0 * c - m1 * d;
  float iv = m0 * d + m1 * c;
  emb[(long)e * 64 + 2 * k]     = (g0 + re + x0) * (1.f / 3.f);
  emb[(long)e * 64 + 2 * k + 1] = (g1 + iv + x1) * (1.f / 3.f);
}

__device__ inline float atomicMaxF(float* addr, float val) {
  if (val >= 0.f)
    return __int_as_float(atomicMax((int*)addr, __float_as_int(val)));
  return __uint_as_float(atomicMin((unsigned int*)addr, __float_as_uint(val)));
}

/* per-edge: a = leaky_relu(a1[seg] + emb @ attn2^T); store a; segment atomic max.
   attn2 (2KB) staged into LDS with CDNA5 async-to-LDS loads (ASYNCcnt). */
__global__ void attn_max_kernel(const float* __restrict__ emb, const float* __restrict__ attn2,
                                const float* __restrict__ a1t, const int* __restrict__ seg,
                                float* __restrict__ a_out, float* __restrict__ m_t) {
  __shared__ float s_attn[HHEADS * DDIM];
  {
    int tid = threadIdx.x;
    if (tid < (HHEADS * DDIM) / 4) {   /* 128 lanes x 16B = 2KB */
      unsigned lds_off = (unsigned)(unsigned long long)&s_attn[tid * 4];
      unsigned long long ga = (unsigned long long)(const void*)(attn2 + tid * 4);
      asm volatile("global_load_async_to_lds_b128 %0, %1, off"
                   :: "v"(lds_off), "v"(ga) : "memory");
    }
    asm volatile("s_wait_asynccnt 0" ::: "memory");
  }
  __syncthreads();
  int e = blockIdx.x * blockDim.x + threadIdx.x;
  if (e >= E_EDGES) return;
  int s = seg[e];
  float acc[HHEADS];
  #pragma unroll
  for (int h = 0; h < HHEADS; ++h) acc[h] = a1t[(long)s * 8 + h];
  const float4* ep = (const float4*)(emb + (long)e * 64);
  #pragma unroll
  for (int i = 0; i < 16; ++i) {
    float4 v = ep[i];
    #pragma unroll
    for (int h = 0; h < HHEADS; ++h) {
      acc[h] += v.x * s_attn[h * DDIM + 4 * i + 0];
      acc[h] += v.y * s_attn[h * DDIM + 4 * i + 1];
      acc[h] += v.z * s_attn[h * DDIM + 4 * i + 2];
      acc[h] += v.w * s_attn[h * DDIM + 4 * i + 3];
    }
  }
  #pragma unroll
  for (int h = 0; h < HHEADS; ++h) {
    float a = acc[h] > 0.f ? acc[h] : LALPHA * acc[h];
    a_out[(long)e * 8 + h] = a;
    atomicMaxF(&m_t[(long)s * 8 + h], a);
  }
}

/* ex = exp(a - m[seg]); store over a; segment sum into den */
__global__ void exp_sum_kernel(const float* __restrict__ m_t, const int* __restrict__ seg,
                               float* __restrict__ a, float* __restrict__ den) {
  long i = (long)blockIdx.x * blockDim.x + threadIdx.x;
  if (i >= (long)E_EDGES * 8) return;
  int e = (int)(i >> 3), h = (int)(i & 7);
  int s = seg[e];
  float ex = expf(a[i] - m_t[(long)s * 8 + h]);
  a[i] = ex;
  atomicAdd(&den[(long)s * 8 + h], ex);
}

/* num[seg, h, :] += (ex/den[seg]) * emb[e, :]; one thread per (edge, dim) */
__global__ void aggregate_kernel(const float* __restrict__ emb, const float* __restrict__ ex,
                                 const float* __restrict__ den, const int* __restrict__ seg,
                                 float* __restrict__ num) {
  long i = (long)blockIdx.x * blockDim.x + threadIdx.x;
  if (i >= (long)E_EDGES * 64) return;
  int e = (int)(i >> 6), j = (int)(i & 63);
  int s = seg[e];
  float ev = emb[(long)e * 64 + j];
  #pragma unroll
  for (int h = 0; h < HHEADS; ++h) {
    float w = ex[(long)e * 8 + h] / den[(long)s * 8 + h];
    atomicAdd(&num[(long)s * 512 + h * 64 + j], w * ev);
  }
}

__global__ void elu_kernel(float* __restrict__ z, long n) {
  long i = (long)blockIdx.x * blockDim.x + threadIdx.x;
  if (i >= n) return;
  float x = z[i];
  z[i] = x > 0.f ? x : (expf(x) - 1.f);
}

__global__ void w1h_kernel(const float* __restrict__ w1, _Float16* __restrict__ w1h) {
  int i = blockIdx.x * blockDim.x + threadIdx.x;
  if (i < HID * FDIM) w1h[i] = (_Float16)w1[i];
}

/* One wave per 16-node tile; wave computes the full 16x128 output tile of
   z @ W1^T with 8 f32 accumulators, so each K-step issues 8 back-to-back
   v_wmma_f32_16x16x32_f16 reusing the same A fragment. Then tanh + W2
   weighting + wave reduction straight to the per-(type,path) score scalar. */
__global__ __launch_bounds__(32) void fusion_score_kernel(
    const float* __restrict__ z0, const float* __restrict__ z1,
    const _Float16* __restrict__ w1h, const float* __restrict__ b1,
    const float* __restrict__ w2, float* __restrict__ scores) {
  const int NTILES = NTOT / 16;                 /* 3750 per path */
  int wid = blockIdx.x;
  int p = wid / NTILES;
  int ntile = wid % NTILES;
  int t, lt;
  if (ntile < NG / 16)                  { t = 0; lt = ntile; }
  else if (ntile < (NG + NM) / 16)      { t = 1; lt = ntile - NG / 16; }
  else                                  { t = 2; lt = ntile - (NG + NM) / 16; }
  const int nodebase = (t == 0) ? 0 : (t == 1 ? NG : NG + NM);
  const float* z = (p == 0 ? z0 : z1) + (long)(nodebase + lt * 16) * FDIM;

  int lane = threadIdx.x;
  int hi = lane >> 4, lid = lane & 15;
  /* A layout: lane<16 holds row lid, K {0..7,16..23}; lane>=16 holds row lid, K {8..15,24..31} */
  const float* arow = z + (long)lid * FDIM + hi * 8;
  const _Float16* bbase = w1h + (long)lid * FDIM + hi * 8;  /* + c*16*FDIM per col tile */

  f8 acc[8];
  #pragma unroll
  for (int c = 0; c < 8; ++c) acc[c] = (f8){0.f, 0.f, 0.f, 0.f, 0.f, 0.f, 0.f, 0.f};

  #pragma unroll 1
  for (int kk = 0; kk < 16; ++kk) {
    int kb = kk * 32;
    half16 a;
    #pragma unroll
    for (int j = 0; j < 8; ++j) {
      a[j]     = (_Float16)arow[kb + j];
      a[j + 8] = (_Float16)arow[kb + 16 + j];
    }
    #pragma unroll
    for (int c = 0; c < 8; ++c) {
      const _Float16* brow = bbase + (long)c * 16 * FDIM;
      half16 b;
      #pragma unroll
      for (int j = 0; j < 8; ++j) {
        b[j]     = brow[kb + j];
        b[j + 8] = brow[kb + 16 + j];
      }
      acc[c] = __builtin_amdgcn_wmma_f32_16x16x32_f16(false, a, false, b, (short)0,
                                                      acc[c], false, false);
    }
  }
  /* C/D layout: VGPR r holds (M = r or 8+r, N = lid) */
  float part = 0.f;
  #pragma unroll
  for (int c = 0; c < 8; ++c) {
    int o = c * 16 + lid;
    float b1v = b1[o], w2v = w2[o];
    #pragma unroll
    for (int r = 0; r < 8; ++r) part += tanhf(acc[c][r] + b1v) * w2v;
  }
  for (int off = 16; off > 0; off >>= 1) part += __shfl_xor(part, off, 32);
  if (lane == 0) atomicAdd(&scores[t * 2 + p], part);
}

/* out = beta0 * z0 + beta1 * z1 (z0 lives in d_out, in-place) */
__global__ void combine_kernel(float* __restrict__ out, const float* __restrict__ z1,
                               const float* __restrict__ scores, int t, float inv_n, long n) {
  long i = (long)blockIdx.x * blockDim.x + threadIdx.x;
  if (i >= n) return;
  float s0 = scores[t * 2 + 0] * inv_n, s1 = scores[t * 2 + 1] * inv_n;
  float mx = fmaxf(s0, s1);
  float e0 = expf(s0 - mx), e1 = expf(s1 - mx);
  float inv = 1.f / (e0 + e1);
  out[i] = (e0 * inv) * out[i] + (e1 * inv) * z1[i];
}

static inline int cdiv(long a, long b) { return (int)((a + b - 1) / b); }

extern "C" void kernel_launch(void* const* d_in, const int* in_sizes, int n_in,
                              void* d_out, int out_size, void* d_ws, size_t ws_size,
                              hipStream_t stream) {
  const float* h_g = (const float*)d_in[0];
  const float* h_m = (const float*)d_in[1];
  const float* h_d = (const float*)d_in[2];
  const int* idx[2][3] = {
      {(const int*)d_in[3], (const int*)d_in[4], (const int*)d_in[5]},
      {(const int*)d_in[6], (const int*)d_in[7], (const int*)d_in[8]}};
  const float* rvec = (const float*)d_in[9];
  const float* attn1p[3] = {(const float*)d_in[10], (const float*)d_in[12],
                            (const float*)d_in[14]};
  const float* attn2p[3] = {(const float*)d_in[11], (const float*)d_in[13],
                            (const float*)d_in[15]};
  const float* W1 = (const float*)d_in[16];
  const float* B1 = (const float*)d_in[17];
  const float* W2 = (const float*)d_in[18];

  float* ws   = (float*)d_ws;
  float* emb  = ws + OFF_EMB;
  float* abuf = ws + OFF_A;
  float* a1b  = ws + OFF_A1;
  float* mbuf = ws + OFF_M;
  float* denb = ws + OFF_DEN;
  _Float16* w1h = (_Float16*)(ws + OFF_W1H);
  float* scores = ws + OFF_SC;
  float* z1   = ws + OFF_Z1;
  float* z0   = (float*)d_out;

  const int nb[3] = {0, NG, NG + NM};
  const long nt[3] = {NG, NM, ND};

  /* init accumulators (every call: graph-replay safe) */
  fill_kernel<<<cdiv(ZTOT, 256), 256, 0, stream>>>(z0, 0.f, ZTOT);
  fill_kernel<<<cdiv(ZTOT, 256), 256, 0, stream>>>(z1, 0.f, ZTOT);
  fill_kernel<<<1, 32, 0, stream>>>(scores, 0.f, 6);

  a1_kernel<<<cdiv((long)NTOT * 8, 256), 256, 0, stream>>>(h_g, h_m, h_d, attn1p[0],
                                                           attn1p[1], attn1p[2], a1b);
  w1h_kernel<<<cdiv((long)HID * FDIM, 256), 256, 0, stream>>>(W1, w1h);

  for (int p = 0; p < 2; ++p) {
    float* zP = p ? z1 : z0;
    semantic_kernel<<<cdiv((long)E_EDGES * 32, 256), 256, 0, stream>>>(
        h_g, h_m, h_d, idx[p][0], idx[p][1], idx[p][2], rvec, p, emb);
    for (int t = 0; t < 3; ++t) {
      float* m_t   = mbuf + (long)nb[t] * 8;
      float* den_t = denb + (long)nb[t] * 8;
      const float* a1t = a1b + (long)nb[t] * 8;
      const int* seg = idx[p][t];
      float* num = zP + (long)nb[t] * 512;
      long nh = nt[t] * 8;
      fill_kernel<<<cdiv(nh, 256), 256, 0, stream>>>(m_t, -INFINITY, nh);
      fill_kernel<<<cdiv(nh, 256), 256, 0, stream>>>(den_t, 0.f, nh);
      attn_max_kernel<<<cdiv(E_EDGES, 256), 256, 0, stream>>>(emb, attn2p[t], a1t, seg,
                                                              abuf, m_t);
      exp_sum_kernel<<<cdiv((long)E_EDGES * 8, 256), 256, 0, stream>>>(m_t, seg, abuf,
                                                                       den_t);
      aggregate_kernel<<<cdiv((long)E_EDGES * 64, 256), 256, 0, stream>>>(emb, abuf,
                                                                          den_t, seg, num);
    }
    elu_kernel<<<cdiv(ZTOT, 256), 256, 0, stream>>>(zP, ZTOT);
  }

  fusion_score_kernel<<<2 * (NTOT / 16), 32, 0, stream>>>(z0, z1, w1h, B1, W2, scores);

  for (int t = 0; t < 3; ++t) {
    long n = nt[t] * 512;
    combine_kernel<<<cdiv(n, 256), 256, 0, stream>>>(z0 + (long)nb[t] * 512,
                                                     z1 + (long)nb[t] * 512, scores, t,
                                                     1.f / (float)nt[t], n);
  }
}
